// BitSelfAttentionHF_11888469475516
// MI455X (gfx1250) — 16-bit and 8-bit matrix operand order repaired and checked by reading, compiled, not run on hardware
//
#include <hip/hip_runtime.h>
#include <hip/hip_bf16.h>

#define BB 8
#define SS 1024
#define DD 1024
#define HH 16
#define DHH 64

typedef __attribute__((ext_vector_type(16))) __bf16 v16bf;
typedef __attribute__((ext_vector_type(4)))  __bf16 v4bf;
typedef __attribute__((ext_vector_type(8)))  float  v8f;
typedef __attribute__((ext_vector_type(4)))  float  v4f;
typedef __attribute__((ext_vector_type(8)))  int    v8i;
typedef __bf16 bf16_t;

__device__ __forceinline__ v8f wmma_bf16(v16bf a, v16bf b, v8f c) {
  return __builtin_amdgcn_wmma_f32_16x16x32_bf16(
      false, a, false, b, (short)0, c, false, false);
}
__device__ __forceinline__ v8i wmma_iu8(v8i a, v8i b, v8i c) {
  return __builtin_amdgcn_wmma_i32_16x16x64_iu8(true, a, true, b, c, false, false);
}

__device__ __forceinline__ v16bf load_a_bf16(const bf16_t* row, int o) {
  v16bf a;
#pragma unroll
  for (int e = 0; e < 16; ++e)
    a[e] = row[(o << 4) + e];
  return a;
}
__device__ __forceinline__ v16bf load_b_bf16(const bf16_t* row, int o) {
  v16bf b;
#pragma unroll
  for (int e = 0; e < 16; ++e) b[e] = row[(o << 4) + e];
  return b;
}
__device__ __forceinline__ v8i load_a_i8(const int* row32, int o) {
  v8i a;
#pragma unroll
  for (int j = 0; j < 4; ++j) {
    a[2 * j]     = row32[4 * j + 2 * o];
    a[2 * j + 1] = row32[4 * j + 2 * o + 1];
  }
  return a;
}
__device__ __forceinline__ v8i load_b_i8(const int* row32, int o) {
  v8i b;
#pragma unroll
  for (int g = 0; g < 4; ++g) b[g] = row32[2 * o + (g & 1) + 4 * (g >> 1)];
#pragma unroll
  for (int g = 0; g < 4; ++g) b[4 + g] = row32[8 + 2 * o + (g & 1) + 4 * (g >> 1)];
  return b;
}

__device__ __forceinline__ float fq8(float x, float sc) {
  float r = rintf(x * sc);
  return fminf(fmaxf(r, -128.f), 127.f) / sc;
}
__device__ __forceinline__ float wave_max(float v) {
#pragma unroll
  for (int m = 16; m > 0; m >>= 1) v = fmaxf(v, __shfl_xor(v, m, 32));
  return v;
}
__device__ __forceinline__ float wave_sum(float v) {
#pragma unroll
  for (int m = 16; m > 0; m >>= 1) v += __shfl_xor(v, m, 32);
  return v;
}
__device__ __forceinline__ float half_max(float v) {
#pragma unroll
  for (int m = 8; m > 0; m >>= 1) v = fmaxf(v, __shfl_xor(v, m, 32));
  return v;
}

__global__ __launch_bounds__(256)
void cvt_bf16_kernel(const float* __restrict__ src, bf16_t* __restrict__ dst,
                     int n4) {
  int i = blockIdx.x * 256 + threadIdx.x;
  if (i >= n4) return;
  v4f v = ((const v4f*)src)[i];
  v4bf o;
  o[0] = (bf16_t)v[0]; o[1] = (bf16_t)v[1];
  o[2] = (bf16_t)v[2]; o[3] = (bf16_t)v[3];
  ((v4bf*)dst)[i] = o;
}

__global__ __launch_bounds__(32)
void qkvproj_wmma(const bf16_t* __restrict__ Xb, const bf16_t* __restrict__ Wb,
                  const float* __restrict__ bq, const float* __restrict__ bk,
                  const float* __restrict__ bv,
                  char* __restrict__ Qi, char* __restrict__ Ki,
                  float* __restrict__ sQ, float* __restrict__ sK,
                  bf16_t* __restrict__ Vt) {
  const int lane = threadIdx.x & 31;
  const int nl   = lane & 15;
  const int o    = lane >> 4;
  const int tt   = blockIdx.x;
  const int h    = blockIdx.y;
  const int mat  = blockIdx.z;

  const bf16_t* W = Wb + (size_t)mat * DD * DD;
  const float* bias = (mat == 0) ? bq : (mat == 1) ? bk : bv;

  const bf16_t* xrow = Xb + (size_t)(tt * 16 + nl) * DD;

  v8f acc[4] = {{}, {}, {}, {}};
  for (int ks = 0; ks < DD / 32; ++ks) {
    v16bf a = load_a_bf16(xrow + ks * 32, o);
#pragma unroll
    for (int nt = 0; nt < 4; ++nt) {
      const bf16_t* wrow = W + (size_t)(h * DHH + nt * 16 + nl) * DD + ks * 32;
      acc[nt] = wmma_bf16(a, load_b_bf16(wrow, o), acc[nt]);
    }
  }

#pragma unroll
  for (int nt = 0; nt < 4; ++nt) {
    float bn = bias[h * DHH + nt * 16 + nl];
#pragma unroll
    for (int r = 0; r < 8; ++r) acc[nt][r] += bn;
  }

#pragma unroll
  for (int r = 0; r < 8; ++r) {
    float am = 0.f;
#pragma unroll
    for (int nt = 0; nt < 4; ++nt) am = fmaxf(am, fabsf(acc[nt][r]));
    am = half_max(am);
    float sc = 127.f / fmaxf(am, 1e-5f);

    int t = tt * 16 + r + 8 * o;
    int b = t >> 10, s = t & (SS - 1);
    size_t tok = ((size_t)b * HH + h) * SS + s;
#pragma unroll
    for (int nt = 0; nt < 4; ++nt) {
      int dl = nt * 16 + nl;
      if (mat == 2) {
        Vt[(((size_t)b * HH + h) * DHH + dl) * SS + s] =
            (bf16_t)fq8(acc[nt][r], sc);
      } else {
        float q = fminf(fmaxf(rintf(acc[nt][r] * sc), -128.f), 127.f);
        char qi = (char)(int)q;
        if (mat == 0) Qi[tok * DHH + dl] = qi;
        else          Ki[tok * DHH + dl] = qi;
      }
    }
    if (mat != 2 && nl == 0) {
      if (mat == 0) sQ[tok] = sc; else sK[tok] = sc;
    }
  }
}

__global__ __launch_bounds__(32)
void attn_wmma(const char* __restrict__ Qi, const char* __restrict__ Ki,
               const float* __restrict__ sQ, const float* __restrict__ sK,
               const bf16_t* __restrict__ Vt, float* __restrict__ out) {
  __shared__ float scb[16 * SS];
  const int lane = threadIdx.x & 31;
  const int nl   = lane & 15;
  const int o    = lane >> 4;
  const int qt   = blockIdx.x;
  const int h    = blockIdx.y;
  const int b    = blockIdx.z;

  const size_t headSD = ((size_t)b * HH + h) * SS;

  const int* qrow32 = (const int*)(Qi + (headSD + qt * 16 + nl) * DHH);
  v8i aQ = load_a_i8(qrow32, o);
  float qinv[8];
#pragma unroll
  for (int r = 0; r < 8; ++r)
    qinv[r] = 0.125f / sQ[headSD + qt * 16 + r + 8 * o];

  for (int kt = 0; kt < SS / 16; ++kt) {
    const int* krow32 = (const int*)(Ki + (headSD + kt * 16 + nl) * DHH);
    v8i acc = {};
    acc = wmma_iu8(aQ, load_b_i8(krow32, o), acc);
    float invk = 1.0f / sK[headSD + kt * 16 + nl];
#pragma unroll
    for (int r = 0; r < 8; ++r)
      scb[(r + 8 * o) * SS + kt * 16 + nl] = (float)acc[r] * qinv[r] * invk;
  }
  __syncthreads();

  bf16_t* pb = (bf16_t*)scb;
  for (int q = 0; q < 16; ++q) {
    float v[32];
    float am = 0.f;
#pragma unroll
    for (int j = 0; j < 32; ++j) {
      v[j] = scb[q * SS + lane + j * 32];
      am = fmaxf(am, fabsf(v[j]));
    }
    am = wave_max(am);
    float s1 = 127.f / fmaxf(am, 1e-5f);

    float mx = -3.4e38f;
#pragma unroll
    for (int j = 0; j < 32; ++j) { v[j] = fq8(v[j], s1); mx = fmaxf(mx, v[j]); }
    mx = wave_max(mx);

    float sum = 0.f;
#pragma unroll
    for (int j = 0; j < 32; ++j) { v[j] = __expf(v[j] - mx); sum += v[j]; }
    sum = wave_sum(sum);
    float inv = 1.f / sum;

    float am2 = 0.f;
#pragma unroll
    for (int j = 0; j < 32; ++j) { v[j] *= inv; am2 = fmaxf(am2, v[j]); }
    am2 = wave_max(am2);
    float s2 = 127.f / fmaxf(am2, 1e-5f);

    __syncthreads();
#pragma unroll
    for (int j = 0; j < 32; ++j)
      pb[q * 2048 + lane + j * 32] = (bf16_t)fq8(v[j], s2);
  }
  __syncthreads();

  const bf16_t* vbase = Vt + ((size_t)b * HH + h) * DHH * SS;
  v8f cacc[4] = {{}, {}, {}, {}};
  for (int ks = 0; ks < SS / 32; ++ks) {
    v16bf a = load_a_bf16(pb + nl * 2048 + ks * 32, o);
#pragma unroll
    for (int nt = 0; nt < 4; ++nt) {
      const bf16_t* vrow = vbase + (size_t)(nt * 16 + nl) * SS + ks * 32;
      cacc[nt] = wmma_bf16(a, load_b_bf16(vrow, o), cacc[nt]);
    }
  }
#pragma unroll
  for (int nt = 0; nt < 4; ++nt) {
#pragma unroll
    for (int r = 0; r < 8; ++r) {
      int s = qt * 16 + r + 8 * o;
      out[((size_t)b * SS + s) * DD + h * DHH + nt * 16 + nl] = cacc[nt][r];
    }
  }
}

extern "C" void kernel_launch(void* const* d_in, const int* in_sizes, int n_in,
                              void* d_out, int out_size, void* d_ws, size_t ws_size,
                              hipStream_t stream) {
  const float* X  = (const float*)d_in[0];
  const float* Wq = (const float*)d_in[1];
  const float* bq = (const float*)d_in[2];
  const float* Wk = (const float*)d_in[3];
  const float* bk = (const float*)d_in[4];
  const float* Wv = (const float*)d_in[5];
  const float* bv = (const float*)d_in[6];
  float* out = (float*)d_out;

  const size_t nX   = (size_t)BB * SS * DD;
  const size_t nW   = (size_t)DD * DD;
  const size_t nTok = (size_t)BB * HH * SS;
  char* p = (char*)d_ws;
  bf16_t* Xb  = (bf16_t*)p;            p += nX * sizeof(bf16_t);
  bf16_t* Wb  = (bf16_t*)p;            p += 3 * nW * sizeof(bf16_t);
  bf16_t* Vt  = (bf16_t*)p;            p += nX * sizeof(bf16_t);
  char*   Qi  = p;                     p += nX;
  char*   Ki  = p;                     p += nX;
  float*  sQs = (float*)p;             p += nTok * sizeof(float);
  float*  sKs = (float*)p;

  cvt_bf16_kernel<<<(int)(nX / 4 / 256), 256, 0, stream>>>(X, Xb, (int)(nX / 4));
  cvt_bf16_kernel<<<(int)(nW / 4 / 256), 256, 0, stream>>>(Wq, Wb, (int)(nW / 4));
  cvt_bf16_kernel<<<(int)(nW / 4 / 256), 256, 0, stream>>>(Wk, Wb + nW, (int)(nW / 4));
  cvt_bf16_kernel<<<(int)(nW / 4 / 256), 256, 0, stream>>>(Wv, Wb + 2 * nW, (int)(nW / 4));

  dim3 g1(BB * SS / 16, HH, 3);
  qkvproj_wmma<<<g1, 32, 0, stream>>>(Xb, Wb, bq, bk, bv, Qi, Ki, sQs, sKs, Vt);

  dim3 g2(SS / 16, HH, BB);
  attn_wmma<<<g2, 32, 0, stream>>>(Qi, Ki, sQs, sKs, Vt, out);
}
